// ContentOnlyRouter_51934744543482
// MI455X (gfx1250) — matrix operand order repaired and checked by reading, not compiled, not run on hardware
//
#include <hip/hip_runtime.h>
#include <hip/hip_bf16.h>
#include <stdint.h>

// Problem sizes (fixed by the reference)
static constexpr int kTOK = 16384;   // B*S = 4*4096
static constexpr int kD   = 1024;
static constexpr int kT   = 8;

typedef __bf16 bf16_t;
typedef __attribute__((ext_vector_type(16))) __bf16 v16bf;
typedef __attribute__((ext_vector_type(8)))  __bf16 v8bf;
typedef __attribute__((ext_vector_type(4)))  __bf16 v4bf;
typedef __attribute__((ext_vector_type(8)))  float  v8f;
typedef __attribute__((ext_vector_type(4)))  float  v4f;

union Frag16 { v16bf v; v8bf h[2]; };

// ---------------------------------------------------------------- utilities
__global__ void k_zero_i32(int* p, int n) {
    int i = threadIdx.x;
    if (i < n) p[i] = 0;
}

// Pack sign(tile_sigs[t][d]) into bit t of signs[d]
__global__ void k_signs(const float* __restrict__ sigs, unsigned* __restrict__ signs) {
    int d = blockIdx.x * 256 + threadIdx.x;
    if (d >= kD) return;
    unsigned m = 0;
#pragma unroll
    for (int t = 0; t < kT; ++t)
        if (sigs[t * kD + d] >= 0.0f) m |= (1u << t);
    signs[d] = m;
}

// Wave-per-token routing: scores[t] = sum_d +-x[d]; argmax -> idx, bucket counts
__global__ void k_route(const float* __restrict__ x, const unsigned* __restrict__ signs,
                        int* __restrict__ idx, int* __restrict__ counts) {
    int wave = threadIdx.x >> 5;
    int lane = threadIdx.x & 31;
    int token = blockIdx.x * 8 + wave;
    const float* xr = x + (size_t)token * kD;

    float s[kT];
#pragma unroll
    for (int t = 0; t < kT; ++t) s[t] = 0.0f;

    for (int i = 0; i < kD / 32; ++i) {
        int d = i * 32 + lane;               // coalesced within the wave
        float xv = xr[d];
        unsigned m = signs[d];
#pragma unroll
        for (int t = 0; t < kT; ++t)
            s[t] += ((m >> t) & 1u) ? xv : -xv;
    }
#pragma unroll
    for (int t = 0; t < kT; ++t) {
#pragma unroll
        for (int o = 16; o > 0; o >>= 1)
            s[t] += __shfl_xor(s[t], o, 32);
    }
    if (lane == 0) {
        int best = 0; float bs = s[0];
#pragma unroll
        for (int t = 1; t < kT; ++t)
            if (s[t] > bs) { bs = s[t]; best = t; }   // first-max tie-break
        idx[token] = best;
        atomicAdd(&counts[best], 1);
    }
}

__global__ void k_scan(const int* __restrict__ counts, int* __restrict__ off,
                       int* __restrict__ cursor) {
    if (threadIdx.x == 0) {
        int o = 0;
        for (int t = 0; t < kT; ++t) { off[t] = o; cursor[t] = o; o += counts[t]; }
        off[kT] = o;
    }
}

__global__ void k_scatter(const int* __restrict__ idx, int* __restrict__ cursor,
                          int* __restrict__ perm) {
    int token = blockIdx.x * 256 + threadIdx.x;
    int t = idx[token];
    int p = atomicAdd(&cursor[t], 1);
    perm[p] = token;
}

// x fp32 -> bf16, 4 elements/thread
__global__ void k_cvt_x(const float* __restrict__ x, bf16_t* __restrict__ xb) {
    size_t i = ((size_t)blockIdx.x * 256 + threadIdx.x) * 4;
    v4f v = *(const v4f*)(x + i);
    v4bf o;
    o.x = (bf16_t)v.x; o.y = (bf16_t)v.y; o.z = (bf16_t)v.z; o.w = (bf16_t)v.w;
    *(v4bf*)(xb + i) = o;
}

// W[t][d][e] fp32 -> Wt[t][e][d] bf16 (32x32 tile transpose through LDS)
__global__ void k_cvt_wt(const float* __restrict__ W, bf16_t* __restrict__ Wt) {
    __shared__ float tile[32][33];
    int t  = blockIdx.z;
    int d0 = blockIdx.x * 32;
    int e0 = blockIdx.y * 32;
    const float* Wp = W + (size_t)t * kD * kD;
    bf16_t* Wo = Wt + (size_t)t * kD * kD;
    int tx = threadIdx.x, ty = threadIdx.y;     // 32 x 8
#pragma unroll
    for (int r = 0; r < 4; ++r)
        tile[ty + 8 * r][tx] = Wp[(size_t)(d0 + ty + 8 * r) * kD + e0 + tx];
    __syncthreads();
#pragma unroll
    for (int r = 0; r < 4; ++r)
        Wo[(size_t)(e0 + ty + 8 * r) * kD + d0 + tx] = (bf16_t)tile[tx][ty + 8 * r];
}

// ---------------------------------------------------------------- main GEMM
// outT = Wt-tile (A, M=16 e-rows) x gathered-token (B, N=16 tokens), K=1024.
// Block: 256 thr = 8 waves = 4 e-groups x 2 token-groups -> 256e x 64tok tile.
// Wave: 64e x 32tok = 4 Mblk x 2 Nblk WMMA accumulators.
// Double-buffered async B staging + A-fragment software pipeline.
static constexpr int kBRowBytes = 272;                  // 128 k * 2B + 16B pad
static constexpr int kBufBytes  = 64 * kBRowBytes;      // 17408 B per buffer

__global__ __launch_bounds__(256) void k_gemm(
    const bf16_t* __restrict__ xb, const bf16_t* __restrict__ Wt,
    const float* __restrict__ bias, const int* __restrict__ off,
    const int* __restrict__ perm, float* __restrict__ out) {

    __shared__ __align__(16) unsigned char smemB[2 * kBufBytes];

    // map blockIdx.x -> (expert t, chunk of 64 tokens within its bucket)
    int c = blockIdx.x;
    int t = -1, lc = 0, acc = 0;
    for (int i = 0; i < kT; ++i) {
        int cnt = off[i + 1] - off[i];
        int nch = (cnt + 63) >> 6;
        if (c < acc + nch) { t = i; lc = c - acc; break; }
        acc += nch;
    }
    if (t < 0) return;
    int base  = off[t] + lc * 64;
    int limit = off[t + 1];

    int tidx = threadIdx.x;
    int lane = tidx & 31;
    int w    = tidx >> 5;
    int eg   = w >> 1;              // 0..3
    int tg   = w & 1;               // 0..1
    int e0   = blockIdx.y * 256 + eg * 64;
    int nlo  = lane & 15;
    int half = lane >> 4;

    // tokens for this wave's two N-blocks (clamped; invalid lanes masked at store)
    int tok[2], val[2];
#pragma unroll
    for (int nb = 0; nb < 2; ++nb) {
        int gs = base + tg * 32 + nb * 16 + nlo;
        val[nb] = (gs < limit);
        tok[nb] = perm[val[nb] ? gs : base];
    }

    // staging tokens (4 x 16 tokens, 16B chunk per lane)
    int stok[4];
#pragma unroll
    for (int sh = 0; sh < 4; ++sh) {
        int gs = base + sh * 16 + (tidx >> 4);
        stok[sh] = perm[gs < limit ? gs : base];
    }

    // accumulators initialized with bias (D row r+8*half -> e = e0+mb*16+8*half+r)
    v8f cacc[4][2];
#pragma unroll
    for (int mb = 0; mb < 4; ++mb) {
#pragma unroll
        for (int r = 0; r < 8; ++r) {
            float bv = bias[t * kD + e0 + mb * 16 + half * 8 + r];
            cacc[mb][0][r] = bv;
            cacc[mb][1][r] = bv;
        }
    }

    const bf16_t* Wtp = Wt + (size_t)t * kD * kD;
    // A base: row e = e0 + mb*16 + nlo ; khalf-dependent 16B chunks
    const bf16_t* abase = Wtp + (size_t)(e0 + nlo) * kD + half * 8;
    // B base in LDS for this lane
    const unsigned char* bbase = smemB + (tg * 32 + nlo) * kBRowBytes + half * 16;

    unsigned ldsStage = ((unsigned)(uintptr_t)smemB) + (unsigned)((tidx & 15) * 16);
    uint64_t gStage = (uint64_t)(uintptr_t)xb + (uint64_t)((tidx & 15) * 16);

    // async-stage 64 token-rows x 128 k (bf16) into LDS buffer `buf`
    auto stageB = [&](int kbi, int buf) {
#pragma unroll
        for (int sh = 0; sh < 4; ++sh) {
            unsigned lds = ldsStage + (unsigned)(buf * kBufBytes)
                         + (unsigned)((sh * 16 + (tidx >> 4)) * kBRowBytes);
            uint64_t ga  = gStage + (uint64_t)stok[sh] * (kD * 2)
                         + (uint64_t)(kbi * 256);            // kb*2 bytes, kb=kbi*128
            asm volatile("global_load_async_to_lds_b128 %0, %1, off"
                         :: "v"(lds), "v"(ga) : "memory");
        }
    };
    // A fragments from global (L2-resident Wt): K = 8h+j and 16+8h+j
    auto loadA = [&](Frag16 (&f)[4], int k0) {
#pragma unroll
        for (int mb = 0; mb < 4; ++mb) {
            const bf16_t* ap = abase + (size_t)(mb * 16) * kD + k0;
            f[mb].h[0] = *(const v8bf*)(ap);
            f[mb].h[1] = *(const v8bf*)(ap + 16);
        }
    };

    Frag16 afr[4], afrN[4];
    loadA(afr, 0);          // prologue: A fragments for k-step 0
    stageB(0, 0);           // prologue: async stage of first B chunk

#pragma unroll
    for (int i = 0; i < 8; ++i) {
        int p = i & 1;
        if (i + 1 < 8) {
            stageB(i + 1, p ^ 1);                           // overlap next copy
            asm volatile("s_wait_asynccnt 4" ::: "memory"); // oldest stage done
        } else {
            asm volatile("s_wait_asynccnt 0" ::: "memory");
        }
        __syncthreads();

        const unsigned char* bb = bbase + p * kBufBytes;
#pragma unroll
        for (int ks = 0; ks < 4; ++ks) {
            int s = i * 4 + ks;
            // B fragments from LDS (K = 8*half + j and 16 + 8*half + j)
            Frag16 bfr[2];
#pragma unroll
            for (int nb = 0; nb < 2; ++nb) {
                const v8bf* bp = (const v8bf*)(bb + nb * 16 * kBRowBytes + ks * 64);
                bfr[nb].h[0] = bp[0];
                bfr[nb].h[1] = bp[2];
            }
            // prefetch A fragments for the next k-step (hide L2 latency)
            if (s + 1 < 32) loadA(afrN, (s + 1) * 32);

#pragma unroll
            for (int mb = 0; mb < 4; ++mb)
#pragma unroll
                for (int nb = 0; nb < 2; ++nb)
                    cacc[mb][nb] = __builtin_amdgcn_wmma_f32_16x16x32_bf16(
                        false, afr[mb].v, false, bfr[nb].v,
                        (short)0, cacc[mb][nb], false, false);

#pragma unroll
            for (int mb = 0; mb < 4; ++mb) afr[mb] = afrN[mb];  // SSA rename
        }
        __syncthreads();
    }

    // store: lane holds 8 consecutive e-values of token tok[nb]
#pragma unroll
    for (int nb = 0; nb < 2; ++nb) {
        if (val[nb]) {
            float* orow = out + (size_t)tok[nb] * kD + e0;
#pragma unroll
            for (int mb = 0; mb < 4; ++mb) {
                v4f lo = { cacc[mb][nb][0], cacc[mb][nb][1], cacc[mb][nb][2], cacc[mb][nb][3] };
                v4f hi = { cacc[mb][nb][4], cacc[mb][nb][5], cacc[mb][nb][6], cacc[mb][nb][7] };
                *(v4f*)(orow + mb * 16 + half * 8)     = lo;
                *(v4f*)(orow + mb * 16 + half * 8 + 4) = hi;
            }
        }
    }
}

// ---------------------------------------------------------------- launcher
extern "C" void kernel_launch(void* const* d_in, const int* in_sizes, int n_in,
                              void* d_out, int out_size, void* d_ws, size_t ws_size,
                              hipStream_t stream) {
    const float* x    = (const float*)d_in[0];   // [4,4096,1024]
    const float* sigs = (const float*)d_in[1];   // [8,1024]
    const float* W    = (const float*)d_in[2];   // [8,1024,1024]
    const float* bias = (const float*)d_in[3];   // [8,1024]
    float* out = (float*)d_out;

    char* ws = (char*)d_ws;
    unsigned* signs  = (unsigned*)(ws + 0);            // 4 KB
    int* idx         = (int*)(ws + 4096);              // 64 KB
    int* perm        = (int*)(ws + 69632);             // 64 KB
    int* counts      = (int*)(ws + 135168);            // 8
    int* off         = (int*)(ws + 135200);            // 9
    int* cursor      = (int*)(ws + 135296);            // 8
    bf16_t* xb       = (bf16_t*)(ws + (1u << 20));               // 32 MB
    bf16_t* Wt       = (bf16_t*)(ws + (1u << 20) + (32u << 20)); // 16 MB

    k_zero_i32<<<1, 32, 0, stream>>>(counts, 8);
    k_zero_i32<<<1, 32, 0, stream>>>(cursor, 8);
    k_signs<<<4, 256, 0, stream>>>(sigs, signs);
    k_route<<<kTOK / 8, 256, 0, stream>>>(x, signs, idx, counts);
    k_scan<<<1, 32, 0, stream>>>(counts, off, cursor);
    k_scatter<<<kTOK / 256, 256, 0, stream>>>(idx, cursor, perm);
    k_cvt_x<<<(kTOK * kD / 4) / 256, 256, 0, stream>>>(x, xb);
    k_cvt_wt<<<dim3(kD / 32, kD / 32, kT), dim3(32, 8), 0, stream>>>(W, Wt);

    // upper bound on total 64-token chunks: 16384/64 + (8-1) = 263 -> 264
    k_gemm<<<dim3(264, kD / 256), 256, 0, stream>>>(xb, Wt, bias, off, perm, out);
}
